// Custom_RNN_48507360641166
// MI455X (gfx1250) — matrix operand order repaired and checked by reading, not compiled, not run on hardware
//
#include <hip/hip_runtime.h>
#include <hip/hip_bf16.h>

// ---------------------------------------------------------------------------
// Custom RNN (B=256, N_IN=1024, N=2048, L=4, T=32, N_OUT=256) on MI455X.
// bf16 WMMA GEMMs (fp32 accum). Per time step the four layer matmuls are
// independent (they use the PREVIOUS step's h) -> one batched GEMM (grid.z =
// layer) + a tiny elementwise sin-chain kernel.
//
// GEMM tiling: block = 8 waves (4 M-groups x 2 N-groups), block tile
// 128M x 128N, wave tile 32M x 64N = 8 f32 accumulators. Per k-step a wave
// issues 12x b128 loads and 8 WMMAs (24 B/lane/WMMA), and the weight panel is
// only read M/128 = 2x per layer per step from the 192 MB L2.
// ---------------------------------------------------------------------------

typedef __bf16 bf16;
typedef __attribute__((ext_vector_type(8)))  __bf16 v8bf;
typedef __attribute__((ext_vector_type(16))) __bf16 v16bf;
typedef __attribute__((ext_vector_type(8)))  float  v8f;

#define B_DIM   256
#define N_IN    1024
#define N_DIM   2048
#define N_OUT   256
#define L_DIM   4
#define T_STEPS 32

union V16U { v16bf v; v8bf h[2]; };

static __device__ __forceinline__ v16bf load16(const bf16* p0, const bf16* p1) {
    V16U u;
    u.h[0] = *(const v8bf*)p0;
    u.h[1] = *(const v8bf*)p1;
    return u.v;
}

// ---------------------------------------------------------------------------
// GEMM: C[z,m,n] = sum_k A[z,m,k] * W[z,n,k] + bias[z,n]   (A @ W^T + bias)
// A, W bf16 row-major (K contiguous); C fp32.
// Requires M % 128 == 0, N % 128 == 0, K % 32 == 0.
// ---------------------------------------------------------------------------
__global__ __launch_bounds__(256)
void gemm_bf16_wmma(const bf16* __restrict__ A, const bf16* __restrict__ W,
                    const float* __restrict__ bias, float* __restrict__ C,
                    int N, int K,
                    long long sA, long long sW, long long sB, long long sC)
{
    const int z = blockIdx.z;
    A    += (size_t)z * sA;
    W    += (size_t)z * sW;
    bias += (size_t)z * sB;
    C    += (size_t)z * sC;

    const int lane = threadIdx.x & 31;
    const int wid  = threadIdx.x >> 5;
    const int wm   = wid & 3;        // 0..3  (M direction, 32 rows each)
    const int wn   = wid >> 2;       // 0..1  (N direction, 64 cols each)
    const int lo   = lane & 15;
    const int hi   = lane >> 4;      // 0 for lanes 0-15, 1 for lanes 16-31

    const int m_base = blockIdx.y * 128 + wm * 32;
    const int n_base = blockIdx.x * 128 + wn * 64;

    v8f acc[2][4] = {{v8f{}, v8f{}, v8f{}, v8f{}},
                     {v8f{}, v8f{}, v8f{}, v8f{}}};

    // Per-lane base pointers following ISA VGPR layouts (CDNA5 ISA §7.12.2).
    const bf16* arow0 = A + (size_t)(m_base + lo) * K;           // M-tile 0
    const bf16* arow1 = arow0 + (size_t)16 * K;                  // M-tile 1
    const bf16* wc0   = W + (size_t)(n_base + 0 * 16 + lo) * K;
    const bf16* wc1   = W + (size_t)(n_base + 1 * 16 + lo) * K;
    const bf16* wc2   = W + (size_t)(n_base + 2 * 16 + lo) * K;
    const bf16* wc3   = W + (size_t)(n_base + 3 * 16 + lo) * K;

    const int a_off = hi * 16;   // lanes 16-31 hold K=16..31 of A
    const int b_off = hi * 16;   // lanes 16-31 hold K=16..31 of B

    for (int k = 0; k < K; k += 32) {
        // A 16x32 frags: lane holds A[row][k+a_off .. k+a_off+15] contiguous.
        v16bf a0 = load16(arow0 + k + a_off, arow0 + k + a_off + 8);
        v16bf a1 = load16(arow1 + k + a_off, arow1 + k + a_off + 8);
        // B 32x16 frags: lane holds W[col][k+b_off .. k+b_off+15] contiguous.
        v16bf b0 = load16(wc0 + k + b_off, wc0 + k + b_off + 8);
        v16bf b1 = load16(wc1 + k + b_off, wc1 + k + b_off + 8);
        v16bf b2 = load16(wc2 + k + b_off, wc2 + k + b_off + 8);
        v16bf b3 = load16(wc3 + k + b_off, wc3 + k + b_off + 8);

        acc[0][0] = __builtin_amdgcn_wmma_f32_16x16x32_bf16(false, a0, false, b0,
                                                            (short)0, acc[0][0], false, false);
        acc[0][1] = __builtin_amdgcn_wmma_f32_16x16x32_bf16(false, a0, false, b1,
                                                            (short)0, acc[0][1], false, false);
        acc[0][2] = __builtin_amdgcn_wmma_f32_16x16x32_bf16(false, a0, false, b2,
                                                            (short)0, acc[0][2], false, false);
        acc[0][3] = __builtin_amdgcn_wmma_f32_16x16x32_bf16(false, a0, false, b3,
                                                            (short)0, acc[0][3], false, false);
        acc[1][0] = __builtin_amdgcn_wmma_f32_16x16x32_bf16(false, a1, false, b0,
                                                            (short)0, acc[1][0], false, false);
        acc[1][1] = __builtin_amdgcn_wmma_f32_16x16x32_bf16(false, a1, false, b1,
                                                            (short)0, acc[1][1], false, false);
        acc[1][2] = __builtin_amdgcn_wmma_f32_16x16x32_bf16(false, a1, false, b2,
                                                            (short)0, acc[1][2], false, false);
        acc[1][3] = __builtin_amdgcn_wmma_f32_16x16x32_bf16(false, a1, false, b3,
                                                            (short)0, acc[1][3], false, false);
    }

    // C/D layout: lane (0-15): N=lo, VGPR r -> M=r; lane (16-31): M=r+8.
#pragma unroll
    for (int i = 0; i < 2; ++i) {
        const int mrow = m_base + i * 16 + hi * 8;
#pragma unroll
        for (int j = 0; j < 4; ++j) {
            const int col = n_base + j * 16 + lo;
            const float bv = bias[col];
#pragma unroll
            for (int r = 0; r < 8; ++r) {
                C[(size_t)(mrow + r) * N + col] = acc[i][j][r] + bv;
            }
        }
    }
}

// fp32 -> bf16 conversion (weights / X), done once per launch into workspace.
__global__ void cvt_f32_bf16(const float* __restrict__ s, bf16* __restrict__ d, int n)
{
    int i = blockIdx.x * blockDim.x + threadIdx.x;
    if (i < n) d[i] = (bf16)s[i];
}

__global__ void zero_bf16(bf16* __restrict__ p, int n)
{
    int i = blockIdx.x * blockDim.x + threadIdx.x;
    if (i < n) p[i] = (bf16)0.0f;
}

// Sequential sin chain for one time step:
//   d = x_in[i]; for l: d = sin(d + m[l][i]); h_bf16[l][i] = d;
__global__ void rnn_chain(const float* __restrict__ xin, const float* __restrict__ m,
                          bf16* __restrict__ h, int ls)
{
    int i = blockIdx.x * blockDim.x + threadIdx.x;
    float d = xin[i];
#pragma unroll
    for (int l = 0; l < L_DIM; ++l) {
        d = sinf(d + m[(size_t)l * ls + i]);
        h[(size_t)l * ls + i] = (bf16)d;
    }
}

extern "C" void kernel_launch(void* const* d_in, const int* in_sizes, int n_in,
                              void* d_out, int out_size, void* d_ws, size_t ws_size,
                              hipStream_t stream)
{
    (void)in_sizes; (void)n_in; (void)out_size; (void)ws_size;

    const float* X       = (const float*)d_in[0];   // (B, N_IN)
    const float* W_in_w  = (const float*)d_in[1];   // (N, N_IN)
    const float* W_in_b  = (const float*)d_in[2];   // (N,)
    const float* W_rec_w = (const float*)d_in[3];   // (L, N, N)
    const float* W_rec_b = (const float*)d_in[4];   // (L, N)
    const float* W_out_w = (const float*)d_in[5];   // (N_OUT, N)
    const float* W_out_b = (const float*)d_in[6];   // (N_OUT,)
    // d_in[7] = T (always 32, compile-time constant here)

    const size_t BN = (size_t)B_DIM * N_DIM;        // 524288

    char* ws = (char*)d_ws;
    float* xin   = (float*)(ws + (0ull  << 20));    //  2 MB  (B,N) fp32
    float* mbuf  = (float*)(ws + (2ull  << 20));    //  8 MB  (L,B,N) fp32
    bf16*  hbuf  = (bf16*) (ws + (10ull << 20));    //  4 MB  (L,B,N) bf16
    bf16*  Xb    = (bf16*) (ws + (14ull << 20));    // 0.5 MB (B,N_IN) bf16
    bf16*  Winb  = (bf16*) (ws + (15ull << 20));    //  4 MB  (N,N_IN) bf16
    bf16*  Wrecb = (bf16*) (ws + (19ull << 20));    // 32 MB  (L,N,N) bf16
    bf16*  Woutb = (bf16*) (ws + (51ull << 20));    //  1 MB  (N_OUT,N) bf16

    const int thr = 256;
    // --- one-time (per launch) conversions to bf16 ---
    cvt_f32_bf16<<<(B_DIM * N_IN) / thr, thr, 0, stream>>>(X, Xb, B_DIM * N_IN);
    cvt_f32_bf16<<<(N_DIM * N_IN) / thr, thr, 0, stream>>>(W_in_w, Winb, N_DIM * N_IN);
    cvt_f32_bf16<<<(L_DIM * N_DIM * N_DIM) / thr, thr, 0, stream>>>(W_rec_w, Wrecb,
                                                                    L_DIM * N_DIM * N_DIM);
    cvt_f32_bf16<<<(N_OUT * N_DIM) / thr, thr, 0, stream>>>(W_out_w, Woutb, N_OUT * N_DIM);
    zero_bf16<<<(int)((L_DIM * BN) / thr), thr, 0, stream>>>(hbuf, (int)(L_DIM * BN));

    // --- input projection: x_in = X @ W_in^T + b_in  (M=256, N=2048, K=1024) ---
    gemm_bf16_wmma<<<dim3(N_DIM / 128, B_DIM / 128, 1), thr, 0, stream>>>(
        Xb, Winb, W_in_b, xin, N_DIM, N_IN, 0, 0, 0, 0);

    // --- T time steps: batched recurrent GEMM (4 layers) + sin chain ---
    for (int t = 0; t < T_STEPS; ++t) {
        gemm_bf16_wmma<<<dim3(N_DIM / 128, B_DIM / 128, L_DIM), thr, 0, stream>>>(
            hbuf, Wrecb, W_rec_b, mbuf, N_DIM, N_DIM,
            (long long)BN, (long long)N_DIM * N_DIM, (long long)N_DIM, (long long)BN);
        rnn_chain<<<(int)(BN / thr), thr, 0, stream>>>(xin, mbuf, hbuf, (int)BN);
    }

    // --- output projection: out = h[L-1] @ W_out^T + b_out (M=256, N=256, K=2048) ---
    gemm_bf16_wmma<<<dim3(N_OUT / 128, B_DIM / 128, 1), thr, 0, stream>>>(
        hbuf + (size_t)(L_DIM - 1) * BN, Woutb, W_out_b, (float*)d_out,
        N_OUT, N_DIM, 0, 0, 0, 0);
}
